// Quantizer_18159121727997
// MI455X (gfx1250) — matrix operand order repaired and checked by reading, not compiled, not run on hardware
//
#include <hip/hip_runtime.h>
#include <hip/hip_bf16.h>

typedef __attribute__((ext_vector_type(16))) __bf16 v16bf;
typedef __attribute__((ext_vector_type(8)))  __bf16 v8bf;
typedef __attribute__((ext_vector_type(8)))  float  v8f;

#define D_DIM 256
#define LDX 264                 // padded LDS row stride (bf16) -> 528B, conflict-free
#define TOK_BLK 64              // tokens per workgroup
#define CODE_BLK 64             // codes per K-tile
#define THREADS 512             // 16 waves (wave32): 4 token tiles x 4 code tiles
#define CHUNKS_PER_THREAD 4     // (CODE_BLK*D_DIM/8) / THREADS  v8bf chunks per plane

__device__ __forceinline__ void split_bf16(float v, __bf16& hi, __bf16& lo) {
    unsigned u  = __builtin_bit_cast(unsigned, v);
    float    fh = __builtin_bit_cast(float, u & 0xFFFF0000u);
    float    fl = v - fh;
    unsigned ul = __builtin_bit_cast(unsigned, fl);
    hi = __builtin_bit_cast(__bf16, (unsigned short)(u  >> 16));
    lo = __builtin_bit_cast(__bf16, (unsigned short)(ul >> 16));
}

// ---- kernel 0: one pass over codebook: bf16 hi/lo planes + c2[k] ----
__global__ void vq_prep_kernel(const float* __restrict__ cb,
                               __bf16* __restrict__ cbh, __bf16* __restrict__ cbl,
                               float* __restrict__ c2) {
    __shared__ float red[D_DIM];
    const int k = blockIdx.x;
    const int d = threadIdx.x;
    float v = cb[(size_t)k * D_DIM + d];
    split_bf16(v, cbh[(size_t)k * D_DIM + d], cbl[(size_t)k * D_DIM + d]);
    red[d] = v * v;
    __syncthreads();
    for (int off = D_DIM / 2; off > 0; off >>= 1) {
        if (d < off) red[d] += red[d + off];
        __syncthreads();
    }
    if (d == 0) c2[k] = red[0];
}

// ---- kernel 1: WMMA distance GEMM + running argmin (double-buffered) ----
// score(t,n) = c2[n] - 2*<x_t,c_n>; xc via bf16x3 (hi*hi + lo*hi + hi*lo)
// Pipeline (1 barrier/iter): barrier -> issue next-tile global loads ->
// WMMA compute from buf p -> ds_store regs to buf 1-p (store targets the
// buffer nobody is reading; next barrier orders it against future readers).
__global__ void vq_argmin_kernel(const float* __restrict__ x,
                                 const __bf16* __restrict__ cbh,
                                 const __bf16* __restrict__ cbl,
                                 const float* __restrict__ c2,
                                 float* __restrict__ outIdxF,
                                 int K) {
    extern __shared__ char smem[];
    __bf16* XH = (__bf16*)smem;                       // [TOK_BLK][LDX]
    __bf16* XL = XH + TOK_BLK * LDX;
    __bf16* Cbase = XL + TOK_BLK * LDX;               // 2 buffers x {CH, CL} planes
    const int PS = CODE_BLK * LDX;                    // plane size (elems)
    float*  SRED = (float*)(Cbase + 4 * PS);          // [TOK_BLK][64]
    int*    IRED = (int*)(SRED + TOK_BLK * 64);       // [TOK_BLK][64]

    const int tid  = threadIdx.x;
    const int lane = tid & 31;
    const int lr   = lane & 15;
    const int h    = lane >> 4;
    const int w    = tid >> 5;        // wave 0..15
    const int tm   = w >> 2;          // token tile 0..3
    const int cn   = w & 3;           // code tile 0..3
    const int t0   = blockIdx.x * TOK_BLK;

    // one-time: stage x block (64 tokens x 256), split f32 -> bf16 hi/lo
    for (int i = tid; i < TOK_BLK * D_DIM; i += THREADS) {
        int r = i >> 8, d = i & 255;
        float v = x[(size_t)(t0 + r) * D_DIM + d];
        split_bf16(v, XH[r * LDX + d], XL[r * LDX + d]);
    }

    // per-lane operand bases; A and B both use contiguous K order k = 16h + s
    const __bf16* pAH = XH + (size_t)(tm * 16 + lr) * LDX + 16 * h;
    const __bf16* pAL = XL + (size_t)(tm * 16 + lr) * LDX + 16 * h;
    const int rowB = (cn * 16 + lr) * LDX + 16 * h;

    // per-thread staging slots (v8bf chunks of a 64x256 bf16 plane)
    int srow[CHUNKS_PER_THREAD], scol[CHUNKS_PER_THREAD];
#pragma unroll
    for (int i = 0; i < CHUNKS_PER_THREAD; ++i) {
        int g = tid + i * THREADS;
        srow[i] = g >> 5;            // 32 chunks of 8 per 256-wide row
        scol[i] = (g & 31) * 8;
    }

    float best[8];
    int   bidx[8];
#pragma unroll
    for (int v = 0; v < 8; ++v) { best[v] = 3.4e38f; bidx[v] = 0; }

    const int kTiles = K / CODE_BLK;
    v8bf hreg[CHUNKS_PER_THREAD], lreg[CHUNKS_PER_THREAD];

    // prologue: load tile 0 into regs, drain into buffer 0 (visibility via
    // the barrier at the top of the first loop iteration)
#pragma unroll
    for (int i = 0; i < CHUNKS_PER_THREAD; ++i) {
        hreg[i] = *(const v8bf*)(cbh + (size_t)srow[i] * D_DIM + scol[i]);
        lreg[i] = *(const v8bf*)(cbl + (size_t)srow[i] * D_DIM + scol[i]);
    }
#pragma unroll
    for (int i = 0; i < CHUNKS_PER_THREAD; ++i) {
        *(v8bf*)(Cbase + srow[i] * LDX + scol[i])      = hreg[i];
        *(v8bf*)(Cbase + PS + srow[i] * LDX + scol[i]) = lreg[i];
    }

    for (int kt = 0; kt < kTiles; ++kt) {
        const int p = kt & 1;
        __syncthreads();   // buf p stores visible; prior readers of buf 1-p done

        // issue next tile's global loads AFTER the barrier so its implicit
        // fence doesn't wait on them; they land during the WMMA phase
        if (kt + 1 < kTiles) {
            const __bf16* srcH = cbh + (size_t)(kt + 1) * CODE_BLK * D_DIM;
            const __bf16* srcL = cbl + (size_t)(kt + 1) * CODE_BLK * D_DIM;
#pragma unroll
            for (int i = 0; i < CHUNKS_PER_THREAD; ++i) {
                hreg[i] = *(const v8bf*)(srcH + (size_t)srow[i] * D_DIM + scol[i]);
                lreg[i] = *(const v8bf*)(srcL + (size_t)srow[i] * D_DIM + scol[i]);
            }
            __builtin_amdgcn_sched_barrier(0);  // keep loads hoisted above compute
        }

        const __bf16* pBH = Cbase + (size_t)(2 * p) * PS + rowB;
        const __bf16* pBL = pBH + PS;
        v8f acc = {0.f, 0.f, 0.f, 0.f, 0.f, 0.f, 0.f, 0.f};
#pragma unroll
        for (int ks = 0; ks < D_DIM / 32; ++ks) {
            const int d0 = ks * 32;
            v16bf ah, al, bh, bl;
            ((v8bf*)&ah)[0] = *(const v8bf*)(pAH + d0);
            ((v8bf*)&ah)[1] = *(const v8bf*)(pAH + d0 + 8);
            ((v8bf*)&al)[0] = *(const v8bf*)(pAL + d0);
            ((v8bf*)&al)[1] = *(const v8bf*)(pAL + d0 + 8);
            ((v8bf*)&bh)[0] = *(const v8bf*)(pBH + d0);
            ((v8bf*)&bh)[1] = *(const v8bf*)(pBH + d0 + 8);
            ((v8bf*)&bl)[0] = *(const v8bf*)(pBL + d0);
            ((v8bf*)&bl)[1] = *(const v8bf*)(pBL + d0 + 8);
            acc = __builtin_amdgcn_wmma_f32_16x16x32_bf16(false, ah, false, bh,
                                                          (short)0, acc, false, false);
            acc = __builtin_amdgcn_wmma_f32_16x16x32_bf16(false, al, false, bh,
                                                          (short)0, acc, false, false);
            acc = __builtin_amdgcn_wmma_f32_16x16x32_bf16(false, ah, false, bl,
                                                          (short)0, acc, false, false);
        }

        const int n = kt * CODE_BLK + cn * 16 + lr;   // this lane's code column
        const float c2n = c2[n];
#pragma unroll
        for (int v = 0; v < 8; ++v) {
            float s = __builtin_fmaf(-2.f, acc[v], c2n);
            if (s < best[v]) { best[v] = s; bidx[v] = n; }  // ascending n keeps first min
        }

        // drain staged regs into the buffer nobody reads this iteration;
        // next iteration's barrier publishes it
        if (kt + 1 < kTiles) {
            __bf16* dstH = Cbase + (size_t)(2 * (1 - p)) * PS;
#pragma unroll
            for (int i = 0; i < CHUNKS_PER_THREAD; ++i) {
                *(v8bf*)(dstH + srow[i] * LDX + scol[i])      = hreg[i];
                *(v8bf*)(dstH + PS + srow[i] * LDX + scol[i]) = lreg[i];
            }
        }
    }

    __syncthreads();
#pragma unroll
    for (int v = 0; v < 8; ++v) {
        int tloc = tm * 16 + v + 8 * h;   // C layout: VGPR v -> row v+8h, col lr
        int slot = cn * 16 + lr;
        SRED[tloc * 64 + slot] = best[v];
        IRED[tloc * 64 + slot] = bidx[v];
    }
    __syncthreads();

    if (tid < TOK_BLK) {
        float bs = 3.4e38f; int bi = 0x7FFFFFFF;
        for (int s = 0; s < 64; ++s) {
            float sc = SRED[tid * 64 + s];
            int   ix = IRED[tid * 64 + s];
            if (sc < bs || (sc == bs && ix < bi)) { bs = sc; bi = ix; }
        }
        outIdxF[t0 + tid] = (float)bi;
    }
}

// ---- kernel 2: gather + per-token squared error ----
__global__ void vq_gather_kernel(const float* __restrict__ x,
                                 const float* __restrict__ cb,
                                 const float* __restrict__ idxF,
                                 float* __restrict__ quant,
                                 float* __restrict__ partial) {
    __shared__ float red[D_DIM];
    const int t = blockIdx.x;
    const int d = threadIdx.x;
    const int k = (int)idxF[t];
    float c  = cb[(size_t)k * D_DIM + d];
    float xv = x[(size_t)t * D_DIM + d];
    quant[(size_t)t * D_DIM + d] = c;   // x + sg(c - x) == c numerically
    float diff = c - xv;
    red[d] = diff * diff;
    __syncthreads();
    for (int off = D_DIM / 2; off > 0; off >>= 1) {
        if (d < off) red[d] += red[d + off];
        __syncthreads();
    }
    if (d == 0) partial[t] = red[0];
}

// ---- kernel 3: deterministic loss reduction ----
__global__ void vq_loss_kernel(const float* __restrict__ partial,
                               float* __restrict__ lossOut, int N, float scale) {
    __shared__ float red[256];
    float s = 0.f;
    for (int i = threadIdx.x; i < N; i += 256) s += partial[i];
    red[threadIdx.x] = s;
    __syncthreads();
    for (int off = 128; off > 0; off >>= 1) {
        if (threadIdx.x < (unsigned)off) red[threadIdx.x] += red[threadIdx.x + off];
        __syncthreads();
    }
    if (threadIdx.x == 0) *lossOut = red[0] * scale;  // codebook + commitment = 2*mean
}

extern "C" void kernel_launch(void* const* d_in, const int* in_sizes, int n_in,
                              void* d_out, int out_size, void* d_ws, size_t ws_size,
                              hipStream_t stream) {
    const float* x  = (const float*)d_in[0];
    const float* cb = (const float*)d_in[1];
    const int N = in_sizes[0] / D_DIM;   // 16384 tokens
    const int K = in_sizes[1] / D_DIM;   // 8192 codes

    float* quant   = (float*)d_out;                 // [N, 256]
    float* idxOut  = quant + (size_t)N * D_DIM;     // [N] (indices as float)
    float* lossOut = idxOut + N;                    // [1]

    // workspace: c2 | partial | codebook bf16 hi plane | lo plane (~8.1 MB)
    float*  c2      = (float*)d_ws;                          // [K]
    float*  partial = c2 + K;                                // [N]
    __bf16* cbh     = (__bf16*)(partial + N);                // [K*256]
    __bf16* cbl     = cbh + (size_t)K * D_DIM;               // [K*256]

    vq_prep_kernel<<<K, D_DIM, 0, stream>>>(cb, cbh, cbl, c2);

    const size_t shBytes = (size_t)(2 * TOK_BLK * LDX + 4 * CODE_BLK * LDX) * sizeof(__bf16)
                         + (size_t)TOK_BLK * 64 * (sizeof(float) + sizeof(int));
    vq_argmin_kernel<<<N / TOK_BLK, THREADS, shBytes, stream>>>(x, cbh, cbl, c2, idxOut, K);

    vq_gather_kernel<<<N, D_DIM, 0, stream>>>(x, cb, idxOut, quant, partial);

    vq_loss_kernel<<<1, 256, 0, stream>>>(partial, lossOut, N,
                                          2.0f / ((float)N * (float)D_DIM));
}
